// CrossAttention_74156905332834
// MI455X (gfx1250) — matrix operand order repaired and checked by reading, not compiled, not run on hardware
//
#include <hip/hip_runtime.h>
#include <hip/hip_bf16.h>
#include <math.h>

typedef __bf16 bf16_t;
typedef __attribute__((ext_vector_type(16))) bf16_t v16bf;
typedef __attribute__((ext_vector_type(8)))  float  v8f;

union FragB {
    v16bf v;
    unsigned int u[8];
    bf16_t h[16];
};

// ---- CDNA5 WMMA fragment layout helpers (16-bit, 16x16x32) -----------------
// A-matrix 16x32: lanes 0-15 -> K 0..15, lanes 16-31 -> K 16..31 (kgrp 0 / 8).
// VGPR v holds K = 2*kgrp + 2v..2v+1.
__device__ __forceinline__ int fragA_koff(int v, int kgrp) {
    return 2 * kgrp + 2 * v;
}

// Load A fragment: rowbase points at the lane's M-row (K contiguous, 4B-aligned).
__device__ __forceinline__ v16bf load_fragA(const bf16_t* rowbase, int kgrp) {
    FragB f;
#pragma unroll
    for (int v = 0; v < 8; ++v)
        f.u[v] = *(const unsigned int*)(rowbase + fragA_koff(v, kgrp));
    return f.v;
}

// B-matrix 32x16: lanes 0-15 -> K 0..15, lanes 16-31 -> K 16..31; N = lane%16.
// colbase points at the lane's N-column stored with K contiguous.
__device__ __forceinline__ v16bf load_fragB(const bf16_t* colbase, int kgrp16) {
    FragB f;
#pragma unroll
    for (int v = 0; v < 8; ++v)
        f.u[v] = *(const unsigned int*)(colbase + kgrp16 + 2 * v);
    return f.v;
}

// ---------------------------------------------------------------------------
// f32 -> bf16 streaming convert
__global__ void cvt_bf16_kernel(const float* __restrict__ in, bf16_t* __restrict__ out, int n) {
    int i = blockIdx.x * blockDim.x + threadIdx.x;
    if (i < n) out[i] = (bf16_t)in[i];
}

// ---------------------------------------------------------------------------
// Tiled bf16 WMMA GEMM: C[M,N] = A[M,K] @ B[K,N] + bias, with epilogue modes:
//   mode 0: bias -> bf16 out
//   mode 1: bias + exact GELU -> bf16 out
//   mode 2: bias -> f32 out
// Block: 256 threads (8 waves), tile 128(M) x 128(N), K-step 32.
// Each wave computes a 32x64 sub-tile (2x4 WMMA tiles -> 8 wmma per K-step).
__global__ __launch_bounds__(256) void gemm_bf16_kernel(
        const bf16_t* __restrict__ A, const bf16_t* __restrict__ Bm,
        const float* __restrict__ bias, void* __restrict__ out,
        int M, int N, int K, int mode) {
    __shared__ bf16_t sA[128 * 34];   // [m][k] padded
    __shared__ bf16_t sB[128 * 34];   // [n][k] padded (transposed on stage)

    const int tid  = threadIdx.x;
    const int lane = tid & 31;
    const int ln16 = lane & 15;
    const int grp  = lane >> 4;
    const int w    = tid >> 5;        // 8 waves
    const int wm   = w & 3;           // 4 along M (32 rows each)
    const int wn   = w >> 2;          // 2 along N (64 cols each)
    const int m0   = blockIdx.y * 128;
    const int n0   = blockIdx.x * 128;

    v8f acc[2][4] = {};

    for (int k0 = 0; k0 < K; k0 += 32) {
        __syncthreads();
        // Stage A tile (128x32 bf16 = 2048 uints), 8 uints per thread.
#pragma unroll
        for (int i = 0; i < 8; ++i) {
            int u  = tid * 8 + i;        // 0..2047
            int m  = u >> 4;             // 16 uints per row
            int kp = u & 15;
            unsigned int val = *(const unsigned int*)(A + (size_t)(m0 + m) * K + k0 + kp * 2);
            *(unsigned int*)(&sA[m * 34 + kp * 2]) = val;
        }
        // Stage B tile transposed: global [k][n] -> LDS [n][k]; 16 elems per thread.
#pragma unroll
        for (int i = 0; i < 16; ++i) {
            int e  = tid * 16 + i;       // 0..4095
            int kk = e >> 7;             // 0..31
            int nn = e & 127;            // coalesced across threads
            sB[nn * 34 + kk] = Bm[(size_t)(k0 + kk) * N + n0 + nn];
        }
        __syncthreads();

        const int kgA = grp ? 8 : 0;
        const int kgB = grp ? 16 : 0;
        v16bf aF[2], bF[4];
#pragma unroll
        for (int i = 0; i < 2; ++i)
            aF[i] = load_fragA(&sA[(wm * 32 + i * 16 + ln16) * 34], kgA);
#pragma unroll
        for (int j = 0; j < 4; ++j)
            bF[j] = load_fragB(&sB[(wn * 64 + j * 16 + ln16) * 34], kgB);
#pragma unroll
        for (int i = 0; i < 2; ++i)
#pragma unroll
            for (int j = 0; j < 4; ++j)
                acc[i][j] = __builtin_amdgcn_wmma_f32_16x16x32_bf16(
                    false, aF[i], false, bF[j], (short)0, acc[i][j], false, false);
    }

    // Epilogue: D layout M = r + 8*grp, N = lane%16.
#pragma unroll
    for (int i = 0; i < 2; ++i)
#pragma unroll
        for (int j = 0; j < 4; ++j) {
            int n = n0 + wn * 64 + j * 16 + ln16;
            float bv_ = bias[n];
#pragma unroll
            for (int r = 0; r < 8; ++r) {
                int m = m0 + wm * 32 + i * 16 + r + 8 * grp;
                float x = acc[i][j][r] + bv_;
                if (mode == 1) x = 0.5f * x * (1.0f + erff(x * 0.70710678118654752f));
                if (mode == 2) ((float*)out)[(size_t)m * N + n] = x;
                else           ((bf16_t*)out)[(size_t)m * N + n] = (bf16_t)x;
            }
        }
}

// ---------------------------------------------------------------------------
// Row layernorm over C=1024, f32 in -> bf16 out, with gain/bias.
__global__ __launch_bounds__(256) void layernorm_kernel(
        const float* __restrict__ in, const float* __restrict__ g,
        const float* __restrict__ b, bf16_t* __restrict__ out) {
    __shared__ float red[256];
    const int row = blockIdx.x;
    const float* x = in + (size_t)row * 1024;
    float vals[4], s = 0.f;
#pragma unroll
    for (int i = 0; i < 4; ++i) { vals[i] = x[threadIdx.x + i * 256]; s += vals[i]; }
    red[threadIdx.x] = s; __syncthreads();
    for (int off = 128; off > 0; off >>= 1) {
        if (threadIdx.x < off) red[threadIdx.x] += red[threadIdx.x + off];
        __syncthreads();
    }
    float mu = red[0] * (1.0f / 1024.0f);
    __syncthreads();
    float vs = 0.f;
#pragma unroll
    for (int i = 0; i < 4; ++i) { float d = vals[i] - mu; vs += d * d; }
    red[threadIdx.x] = vs; __syncthreads();
    for (int off = 128; off > 0; off >>= 1) {
        if (threadIdx.x < off) red[threadIdx.x] += red[threadIdx.x + off];
        __syncthreads();
    }
    float rstd = rsqrtf(red[0] * (1.0f / 1024.0f) + 1e-5f);
#pragma unroll
    for (int i = 0; i < 4; ++i) {
        int c = threadIdx.x + i * 256;
        out[(size_t)row * 1024 + c] = (bf16_t)((vals[i] - mu) * rstd * g[c] + b[c]);
    }
}

// ---------------------------------------------------------------------------
// Flash cross-attention, all WMMA bf16. One wave = 16 query rows of one (b,h).
// Scores computed transposed (A=K-tile, B=Q^T) so softmax stats are per-lane;
// y accumulated transposed (A=V^T, B=P), rescale is a per-lane scalar mul.
__global__ __launch_bounds__(128) void attention_kernel(
        const bf16_t* __restrict__ q, const bf16_t* __restrict__ k,
        const bf16_t* __restrict__ v, const int* __restrict__ mask,
        bf16_t* __restrict__ y) {
    __shared__ int    smask[1024];
    __shared__ bf16_t sp[4][16 * 34];   // per-wave P tile [t][s] (16 x 32, padded)

    const int b    = blockIdx.z;
    const int h    = blockIdx.y;
    const int tid  = threadIdx.x;
    const int w    = tid >> 5;
    const int lane = tid & 31;
    const int ln16 = lane & 15;
    const int grp  = lane >> 4;
    const int t0   = blockIdx.x * 64 + w * 16;
    const int C = 1024, D = 64, T = 2048, S = 1024;

    for (int i = tid; i < S; i += 128) smask[i] = mask[b * S + i];
    __syncthreads();

    // Preload Q^T B-fragments for both K(=d)-chunks: N = t = t0 + ln16.
    const size_t qrow = ((size_t)(b * T) + t0 + ln16) * C + h * D;
    v16bf qB0, qB1;
    {
        FragB f0, f1;
        const int kg = grp ? 16 : 0;
#pragma unroll
        for (int vv = 0; vv < 8; ++vv) {
            f0.u[vv] = *(const unsigned int*)(q + qrow + kg + 2 * vv);
            f1.u[vv] = *(const unsigned int*)(q + qrow + 32 + kg + 2 * vv);
        }
        qB0 = f0.v; qB1 = f1.v;
    }

    float m_i = -INFINITY, l_i = 0.f;
    v8f yacc[4] = {};
    const float scale = 0.125f;          // 1/sqrt(64)
    bf16_t* myp = sp[w];

    for (int sb = 0; sb < S; sb += 32) {
        // Prefetch next s-block of K/V into cache (gfx1250 global_prefetch_b8).
        if (sb + 32 < S) {
            __builtin_prefetch(k + ((size_t)(b * S) + sb + 32 + ln16) * C + h * D, 0, 1);
            __builtin_prefetch(v + ((size_t)(b * S) + sb + 32 + ln16) * C + h * D, 0, 1);
        }

        float pval[16];
        float mloc = -INFINITY;
#pragma unroll
        for (int half = 0; half < 2; ++half) {
            const int srow = sb + half * 16 + ln16;                 // A-row: M = s
            const bf16_t* kbase = k + ((size_t)(b * S) + srow) * C + h * D;
            const int kg = grp ? 8 : 0;
            v16bf kA0 = load_fragA(kbase, kg);
            v16bf kA1 = load_fragA(kbase + 32, kg);
            v8f st = {};
            st = __builtin_amdgcn_wmma_f32_16x16x32_bf16(false, kA0, false, qB0, (short)0, st, false, false);
            st = __builtin_amdgcn_wmma_f32_16x16x32_bf16(false, kA1, false, qB1, (short)0, st, false, false);
#pragma unroll
            for (int r = 0; r < 8; ++r) {
                int sg = sb + half * 16 + r + 8 * grp;              // D layout: M = r+8*grp
                float x = st[r] * scale;
                if (smask[sg] == 0) x = -INFINITY;
                pval[half * 8 + r] = x;
                mloc = fmaxf(mloc, x);
            }
        }
        mloc = fmaxf(mloc, __shfl_xor(mloc, 16, 32));
        float mnew  = fmaxf(m_i, mloc);
        float alpha = (mnew == -INFINITY) ? 1.0f : __expf(m_i - mnew);
        float rsum = 0.f;
#pragma unroll
        for (int e = 0; e < 16; ++e) {
            float p = (mnew == -INFINITY) ? 0.f : __expf(pval[e] - mnew);
            pval[e] = p;
            rsum += p;
        }
        rsum += __shfl_xor(rsum, 16, 32);
        l_i = l_i * alpha + rsum;
        m_i = mnew;

        // Store P to LDS as [t][s_local]: t = ln16 (N of score tile).
#pragma unroll
        for (int half = 0; half < 2; ++half)
#pragma unroll
            for (int r = 0; r < 8; ++r)
                myp[ln16 * 34 + half * 16 + r + 8 * grp] = (bf16_t)pval[half * 8 + r];

        // Rescale transposed-y accumulators (per-lane: t = ln16).
#pragma unroll
        for (int dt = 0; dt < 4; ++dt)
#pragma unroll
            for (int r = 0; r < 8; ++r) yacc[dt][r] *= alpha;

        // P as B-fragment (K = s_local 0..31, N = t).
        v16bf pB = load_fragB(&myp[ln16 * 34], grp ? 16 : 0);

        // V^T as A-fragments (M = d, K = s), accumulate y^T.
#pragma unroll
        for (int dt = 0; dt < 4; ++dt) {
            FragB f;
            const int col = h * D + dt * 16 + ln16;
            const int kg  = grp ? 8 : 0;
#pragma unroll
            for (int e = 0; e < 16; ++e) {
                int sl = 2 * kg + e;
                f.h[e] = v[((size_t)(b * S) + sb + sl) * C + col];
            }
            yacc[dt] = __builtin_amdgcn_wmma_f32_16x16x32_bf16(
                false, f.v, false, pB, (short)0, yacc[dt], false, false);
        }
    }

    // y^T layout: M = d = dt*16 + r + 8*grp, N = t = ln16.
    const float invl = 1.0f / l_i;
    const int tg = t0 + ln16;
#pragma unroll
    for (int dt = 0; dt < 4; ++dt)
#pragma unroll
        for (int r = 0; r < 8; ++r) {
            int d = dt * 16 + r + 8 * grp;
            y[((size_t)(b * T) + tg) * C + h * D + d] = (bf16_t)(yacc[dt][r] * invl);
        }
}

// ---------------------------------------------------------------------------
extern "C" void kernel_launch(void* const* d_in, const int* in_sizes, int n_in,
                              void* d_out, int out_size, void* d_ws, size_t ws_size,
                              hipStream_t stream) {
    (void)in_sizes; (void)n_in; (void)out_size; (void)ws_size;
    const int B = 4, T = 2048, S = 1024, C = 1024, E = 512;

    const float* x   = (const float*)d_in[0];
    const float* ee  = (const float*)d_in[1];
    const int*   msk = (const int*)  d_in[2];
    const float* W1  = (const float*)d_in[3];  const float* b1 = (const float*)d_in[4];
    const float* W2  = (const float*)d_in[5];  const float* b2 = (const float*)d_in[6];
    const float* lng = (const float*)d_in[7];  const float* lnb = (const float*)d_in[8];
    const float* Wq  = (const float*)d_in[9];  const float* bq = (const float*)d_in[10];
    const float* Wk  = (const float*)d_in[11]; const float* bk = (const float*)d_in[12];
    const float* Wv  = (const float*)d_in[13]; const float* bv = (const float*)d_in[14];
    const float* Wp  = (const float*)d_in[15]; const float* bp = (const float*)d_in[16];

    char* ws = (char*)d_ws;
    const size_t MB = 1024ull * 1024ull;
    bf16_t* W1b  = (bf16_t*)(ws + 0 * MB);    // 1 MB
    bf16_t* W2b  = (bf16_t*)(ws + 1 * MB);    // 2 MB
    bf16_t* Wqb  = (bf16_t*)(ws + 3 * MB);    // 2 MB
    bf16_t* Wkb  = (bf16_t*)(ws + 5 * MB);    // 2 MB
    bf16_t* Wvb  = (bf16_t*)(ws + 7 * MB);    // 2 MB
    bf16_t* Wpb  = (bf16_t*)(ws + 9 * MB);    // 2 MB
    bf16_t* eeb  = (bf16_t*)(ws + 11 * MB);   // 4 MB  (B*S*E)
    bf16_t* xb   = (bf16_t*)(ws + 15 * MB);   // 16 MB (B*T*C)
    bf16_t* hb   = (bf16_t*)(ws + 31 * MB);   // 8 MB  (B*S*C)
    float*  tmpf = (float*) (ws + 39 * MB);   // 16 MB (B*S*C f32)
    bf16_t* encb = (bf16_t*)(ws + 55 * MB);   // 8 MB
    bf16_t* qb   = (bf16_t*)(ws + 63 * MB);   // 16 MB
    bf16_t* kb   = (bf16_t*)(ws + 79 * MB);   // 8 MB
    bf16_t* vb   = (bf16_t*)(ws + 87 * MB);   // 8 MB
    bf16_t* yb   = (bf16_t*)(ws + 95 * MB);   // 16 MB

    auto cvt = [&](const float* src, bf16_t* dst, int n) {
        cvt_bf16_kernel<<<(n + 255) / 256, 256, 0, stream>>>(src, dst, n);
    };
    cvt(ee, eeb, B * S * E);
    cvt(x,  xb,  B * T * C);
    cvt(W1, W1b, E * C);
    cvt(W2, W2b, C * C);
    cvt(Wq, Wqb, C * C);
    cvt(Wk, Wkb, C * C);
    cvt(Wv, Wvb, C * C);
    cvt(Wp, Wpb, C * C);

    // h = gelu(ee @ W1 + b1)            M=4096, K=512, N=1024
    gemm_bf16_kernel<<<dim3(C / 128, (B * S) / 128), 256, 0, stream>>>(
        eeb, W1b, b1, hb, B * S, C, E, 1);
    // tmp = h @ W2 + b2 (f32)           M=4096, K=1024, N=1024
    gemm_bf16_kernel<<<dim3(C / 128, (B * S) / 128), 256, 0, stream>>>(
        hb, W2b, b2, tmpf, B * S, C, C, 2);
    // enc = layernorm(tmp) -> bf16
    layernorm_kernel<<<B * S, 256, 0, stream>>>(tmpf, lng, lnb, encb);
    // q = x @ Wq + bq                   M=8192
    gemm_bf16_kernel<<<dim3(C / 128, (B * T) / 128), 256, 0, stream>>>(
        xb, Wqb, bq, qb, B * T, C, C, 0);
    // k = enc @ Wk + bk ; v = enc @ Wv + bv
    gemm_bf16_kernel<<<dim3(C / 128, (B * S) / 128), 256, 0, stream>>>(
        encb, Wkb, bk, kb, B * S, C, C, 0);
    gemm_bf16_kernel<<<dim3(C / 128, (B * S) / 128), 256, 0, stream>>>(
        encb, Wvb, bv, vb, B * S, C, C, 0);
    // flash attention -> y (bf16)
    attention_kernel<<<dim3(T / 64, 16, B), 128, 0, stream>>>(qb, kb, vb, msk, yb);
    // out = y @ Wp + bp (f32)
    gemm_bf16_kernel<<<dim3(C / 128, (B * T) / 128), 256, 0, stream>>>(
        yb, Wpb, bp, (float*)d_out, B * T, C, C, 2);
}
